// MambaBiLSTM_91250875171203
// MI455X (gfx1250) — matrix operand order repaired and checked by reading, not compiled, not run on hardware
//
#include <hip/hip_runtime.h>
#include <hip/hip_fp16.h>
#include <math.h>

// ---------------------------------------------------------------------------
// MambaBiLSTM for MI455X (gfx1250)
//  - f16-packed WMMA GEMMs (v_wmma_f32_16x16x32_f16), 16x64 strip per wave
//  - weights pre-converted to f16 once per launch (resident in 192MB L2)
//  - LDS-staged selective scan + 1024-thread LSTM recurrence
// ---------------------------------------------------------------------------

typedef __attribute__((ext_vector_type(16))) _Float16 v16h;
typedef __attribute__((ext_vector_type(8)))  _Float16 v8h;
typedef __attribute__((ext_vector_type(4)))  _Float16 v4h;
typedef __attribute__((ext_vector_type(8)))  float    v8f;

#define Bsz 4
#define Ls  512
#define Dm  512
#define HIDc 512
#define H2c 256
#define NLc 3
#define DSTc 16
#define DCVc 4
#define DINc 1024
#define DTRc 32
#define BL  (Bsz * Ls)   // 2048 rows for every GEMM

__device__ __forceinline__ float sigmoidf_(float x) {
    return 1.0f / (1.0f + __expf(-x));
}

// ---------------------------------------------------------------------------
// f32 -> packed f16 converter. src rows have stride srcStride, dst is [rows,K]
// packed. 4 elements per thread (K % 4 == 0 everywhere here).
// ---------------------------------------------------------------------------
__global__ void cvt_f16_kernel(const float* __restrict__ src, int srcStride,
                               _Float16* __restrict__ dst, int K, long total)
{
    const long e = ((long)blockIdx.x * blockDim.x + threadIdx.x) * 4;
    if (e >= total) return;
    const long row = e / K;
    const int  col = (int)(e % K);
    const float4 v = *(const float4*)(src + row * (size_t)srcStride + col);
    v4h h;
    h[0] = (_Float16)v.x; h[1] = (_Float16)v.y;
    h[2] = (_Float16)v.z; h[3] = (_Float16)v.w;
    *(v4h*)(dst + e) = h;
}

// ---------------------------------------------------------------------------
// WMMA GEMM:  C[M,N] = act( A16[M,K] * W16[N,K]^T + bias0 + bias1 )
// One wave computes a 16x64 strip (4 accumulators, 4 v_wmma per K-step, A
// fragment reused 4x). Operands are packed f16 so every fragment is a
// contiguous b128 load.
// Fragment layouts per CDNA5 ISA 7.12.2 (wave32):
//   A 16x32 f16 : lane m = lane&15 ; K halves {0..7,16..23} (lanes 0-15)
//                                     / {8..15,24..31}      (lanes 16-31)
//   B 32x16 f16 : lane n = lane&15 ; this code loads A and B both as K 0..15 (lanes 0-15) / 16..31 (16-31)
//   C/D 16x16 f32: col = lane&15, VGPR r -> row r + 8*(lane>=16)
// ---------------------------------------------------------------------------
#define ACT_NONE 0
#define ACT_SOFTPLUS 1

template <int ACT>
__global__ __launch_bounds__(256)
void gemm_wmma_f16_kernel(const _Float16* __restrict__ A16, // [M,K] packed
                          const _Float16* __restrict__ W16, // [N,K] packed
                          const float* __restrict__ bias0,  // nullable, [N]
                          const float* __restrict__ bias1,  // nullable, [N]
                          float* __restrict__ C,
                          int M, int N, int K)
{
    const int lane    = threadIdx.x & 31;
    const int wave    = threadIdx.x >> 5;
    const int stripsN = N >> 6;                 // strips of 4 x 16 cols
    const int stripId = blockIdx.x * 8 + wave;
    if (stripId >= (M >> 4) * stripsN) return;  // wave-uniform exit, EXEC all-1s

    const int mt = stripId / stripsN;
    const int ns = stripId % stripsN;
    const int lo = lane & 15;
    const int hi = lane >> 4;                   // 0 or 1

    const _Float16* __restrict__ arow  = A16 + (size_t)((mt << 4) + lo) * K + hi * 16;
    const _Float16* __restrict__ wbase = W16 + (size_t)((ns << 6) + lo) * K + hi * 16;

    v8f acc[4] = {};
    for (int k0 = 0; k0 < K; k0 += 32) {
        const v8h alo = *(const v8h*)(arow + k0);        // K = hi*16 + 0..7
        const v8h ahi = *(const v8h*)(arow + k0 + 8);    // K = hi*16 + 8..15
        const v16h af = __builtin_shufflevector(alo, ahi,
                            0, 1, 2, 3, 4, 5, 6, 7, 8, 9, 10, 11, 12, 13, 14, 15);
#pragma unroll
        for (int t = 0; t < 4; ++t) {
            const v16h bf = *(const v16h*)(wbase + (size_t)(t << 4) * K + k0);
            acc[t] = __builtin_amdgcn_wmma_f32_16x16x32_f16(
                         false, af, false, bf, (short)0, acc[t], false, false);
        }
    }

    const int mbase = mt << 4;
#pragma unroll
    for (int t = 0; t < 4; ++t) {
        const int ncol = (ns << 6) + (t << 4) + lo;
        float bn = 0.0f;
        if (bias0) bn += bias0[ncol];
        if (bias1) bn += bias1[ncol];
#pragma unroll
        for (int r = 0; r < 8; ++r) {
            const int mg = mbase + r + (hi ? 8 : 0);
            float v = acc[t][r] + bn;
            if (ACT == ACT_SOFTPLUS)
                v = (v > 20.0f) ? v : log1pf(__expf(v));
            C[(size_t)mg * N + ncol] = v;
        }
    }
}

// ---------------------------------------------------------------------------
// Depthwise causal conv (width 4) over L + bias + SiLU.
// ---------------------------------------------------------------------------
__global__ void conv_silu_kernel(const float* __restrict__ xz,
                                 const float* __restrict__ cw,  // [DIN,4]
                                 const float* __restrict__ cb,  // [DIN]
                                 float* __restrict__ u)
{
    const int id = blockIdx.x * blockDim.x + threadIdx.x;
    if (id >= Bsz * Ls * DINc) return;
    const int c = id % DINc;
    const int l = (id / DINc) % Ls;
    const int b = id / (DINc * Ls);

    float acc = cb[c];
#pragma unroll
    for (int k = 0; k < DCVc; ++k) {
        const int ls = l + k - (DCVc - 1);
        if (ls >= 0)
            acc += cw[c * DCVc + k] * xz[((size_t)(b * Ls + ls)) * (2 * DINc) + c];
    }
    u[id] = acc * sigmoidf_(acc);
}

// ---------------------------------------------------------------------------
// Selective scan. One thread per (b, channel). 16-wide SSM state in registers,
// per-timestep B/C vectors (32 floats) staged in LDS, shared by 256 channels.
// Fuses  y = scan + u*Dp  and  y *= silu(z).
// ---------------------------------------------------------------------------
__global__ __launch_bounds__(256)
void scan_kernel(const float* __restrict__ dt,   // [B,L,DIN]
                 const float* __restrict__ u,    // [B,L,DIN]
                 const float* __restrict__ xdbl, // [B,L,64] cols 32:48=B, 48:64=C
                 const float* __restrict__ xz,   // [B,L,2*DIN] (z = cols DIN:)
                 const float* __restrict__ Alog, // [DIN,16]
                 const float* __restrict__ Dp,   // [DIN]
                 float* __restrict__ yout)       // [B,L,DIN]
{
    __shared__ float sBC[32];
    const int b = blockIdx.x >> 2;
    const int c = ((blockIdx.x & 3) << 8) + threadIdx.x;

    float Ac[DSTc], h[DSTc];
#pragma unroll
    for (int n = 0; n < DSTc; ++n) {
        Ac[n] = -__expf(Alog[(size_t)c * DSTc + n]);
        h[n]  = 0.0f;
    }
    const float dpc = Dp[c];

    for (int l = 0; l < Ls; ++l) {
        const size_t row = (size_t)(b * Ls + l);
        if (threadIdx.x < 32)
            sBC[threadIdx.x] = xdbl[row * 64 + DTRc + threadIdx.x];
        __syncthreads();

        const float dt_t = dt[row * DINc + c];
        const float u_t  = u[row * DINc + c];
        const float z    = xz[row * (2 * DINc) + DINc + c];
        const float du   = dt_t * u_t;

        float acc = 0.0f;
#pragma unroll
        for (int n = 0; n < DSTc; ++n) {
            h[n] = __expf(dt_t * Ac[n]) * h[n] + du * sBC[n];
            acc += h[n] * sBC[DSTc + n];
        }
        const float y = acc + u_t * dpc;
        yout[row * DINc + c] = y * (z * sigmoidf_(z));
        __syncthreads();
    }
}

// ---------------------------------------------------------------------------
// m = LayerNorm(m + mo) * g + b  (one 256-thread block per (b,l) row, D=512)
// ---------------------------------------------------------------------------
__global__ __launch_bounds__(256)
void add_ln_kernel(float* __restrict__ m, const float* __restrict__ mo,
                   const float* __restrict__ g, const float* __restrict__ bb)
{
    __shared__ float s1[256], s2[256];
    const size_t row = blockIdx.x;
    const int t = threadIdx.x;

    const float v0 = m[row * Dm + t]       + mo[row * Dm + t];
    const float v1 = m[row * Dm + t + 256] + mo[row * Dm + t + 256];
    s1[t] = v0 + v1;
    s2[t] = v0 * v0 + v1 * v1;
    __syncthreads();
    for (int off = 128; off > 0; off >>= 1) {
        if (t < off) { s1[t] += s1[t + off]; s2[t] += s2[t + off]; }
        __syncthreads();
    }
    const float mean = s1[0] * (1.0f / Dm);
    const float var  = s2[0] * (1.0f / Dm) - mean * mean;
    const float inv  = rsqrtf(var + 1e-5f);

    m[row * Dm + t]       = (v0 - mean) * inv * g[t]       + bb[t];
    m[row * Dm + t + 256] = (v1 - mean) * inv * g[t + 256] + bb[t + 256];
}

// ---------------------------------------------------------------------------
// LSTM recurrence, 1024 threads: thread = (gate q, unit j), all 4 batches.
// Gates exchanged through LDS; h state in LDS, c state in q==0 registers.
// Serial chain per step is 256 MACs (vs 1024 in a 256-thread version).
// ---------------------------------------------------------------------------
__global__ __launch_bounds__(1024)
void lstm_kernel(const float* __restrict__ gproj,   // [2,B,L,4*H2]
                 const float* __restrict__ whh_all, // [2,4*H2,H2]
                 float* __restrict__ lf,            // [B,L,H2]
                 float* __restrict__ lb)            // [B,L,H2]
{
    __shared__ float hS[Bsz][H2c];          //  4 KB
    __shared__ float gS[4][Bsz][H2c];       // 16 KB
    const int dir = blockIdx.x;
    const int q   = threadIdx.x >> 8;
    const int j   = threadIdx.x & 255;

    const float* __restrict__ gp  = gproj   + (size_t)dir * Bsz * Ls * (4 * H2c);
    const float* __restrict__ whh = whh_all + (size_t)dir * (4 * H2c) * H2c;
    float* __restrict__ out = dir ? lb : lf;

    float cr[Bsz];
    if (q == 0) {
#pragma unroll
        for (int b = 0; b < Bsz; ++b) { cr[b] = 0.0f; hS[b][j] = 0.0f; }
    }
    __syncthreads();

    const float* __restrict__ wr = whh + (size_t)(q * H2c + j) * H2c;
    for (int s = 0; s < Ls; ++s) {
        const int t = dir ? (Ls - 1 - s) : s;
        float a[Bsz];
#pragma unroll
        for (int b = 0; b < Bsz; ++b)
            a[b] = gp[((size_t)(b * Ls + t)) * (4 * H2c) + q * H2c + j];
        for (int kk = 0; kk < H2c; ++kk) {
            const float w = wr[kk];
#pragma unroll
            for (int b = 0; b < Bsz; ++b) a[b] += w * hS[b][kk];
        }
#pragma unroll
        for (int b = 0; b < Bsz; ++b) gS[q][b][j] = a[b];
        __syncthreads();                      // hS reads + gS writes done
        if (q == 0) {
#pragma unroll
            for (int b = 0; b < Bsz; ++b) {
                const float ig = sigmoidf_(gS[0][b][j]);
                const float fg = sigmoidf_(gS[1][b][j]);
                const float gg = tanhf(gS[2][b][j]);
                const float og = sigmoidf_(gS[3][b][j]);
                const float c  = fg * cr[b] + ig * gg;
                const float hh = og * tanhf(c);
                cr[b] = c;
                hS[b][j] = hh;
                out[((size_t)(b * Ls + t)) * H2c + j] = hh;
            }
        }
        __syncthreads();                      // new hS visible to all
    }
}

// ---------------------------------------------------------------------------
// comb = concat([m (512), lf (256), lb (256)], axis=-1)
// ---------------------------------------------------------------------------
__global__ void concat_kernel(const float* __restrict__ m,
                              const float* __restrict__ lf,
                              const float* __restrict__ lb,
                              float* __restrict__ comb)
{
    const int id = blockIdx.x * blockDim.x + threadIdx.x;
    if (id >= BL * (Dm + 2 * H2c)) return;
    const int j   = id % (Dm + 2 * H2c);
    const int row = id / (Dm + 2 * H2c);
    float v;
    if (j < Dm)            v = m[(size_t)row * Dm + j];
    else if (j < Dm + H2c) v = lf[(size_t)row * H2c + (j - Dm)];
    else                   v = lb[(size_t)row * H2c + (j - Dm - H2c)];
    comb[id] = v;
}

// ---------------------------------------------------------------------------
extern "C" void kernel_launch(void* const* d_in, const int* in_sizes, int n_in,
                              void* d_out, int out_size, void* d_ws, size_t ws_size,
                              hipStream_t stream)
{
    (void)in_sizes; (void)n_in; (void)out_size; (void)ws_size;

    const float* x      = (const float*)d_in[0];
    const float* w_in   = (const float*)d_in[1];   // [3, 2048, 512]
    const float* conv_w = (const float*)d_in[2];   // [3, 1024, 4]
    const float* conv_b = (const float*)d_in[3];   // [3, 1024]
    const float* xpw    = (const float*)d_in[4];   // [3, 64, 1024]
    const float* dtw    = (const float*)d_in[5];   // [3, 1024, 32]
    const float* dtbias = (const float*)d_in[6];   // [3, 1024]
    const float* Alog   = (const float*)d_in[7];   // [3, 1024, 16]
    const float* Dp     = (const float*)d_in[8];   // [3, 1024]
    const float* wo     = (const float*)d_in[9];   // [3, 512, 1024]
    const float* lng    = (const float*)d_in[10];  // [3, 512]
    const float* lnb    = (const float*)d_in[11];  // [3, 512]
    const float* wih    = (const float*)d_in[12];  // [2, 1024, 512]
    const float* whh    = (const float*)d_in[13];  // [2, 1024, 256]
    const float* bih    = (const float*)d_in[14];  // [2, 1024]
    const float* bhh    = (const float*)d_in[15];  // [2, 1024]
    const float* fw     = (const float*)d_in[16];  // [512, 1024]
    const float* fb     = (const float*)d_in[17];  // [512]

    // ------------------- workspace layout -------------------
    float* ws   = (float*)d_ws;
    size_t off  = 0;
    float* mbuf = ws + off; off += (size_t)BL * Dm;        // residual stream
    float* xzb  = ws + off; off += (size_t)BL * 2 * DINc;  // in_proj out
    float* ubuf = ws + off; off += (size_t)BL * DINc;      // conv+silu out
    float* xdbl = ws + off; off += (size_t)BL * 64;        // x_proj out
    float* dtb  = ws + off; off += (size_t)BL * DINc;      // softplus(dt)
    float* ybuf = ws + off; off += (size_t)BL * DINc;      // scan out (gated)
    float* mob  = ws + off; off += (size_t)BL * Dm;        // out_proj out

    // f16 arenas after the f32 region
    _Float16* h16   = (_Float16*)(ws + off);
    size_t ho = 0;
    _Float16* w16in = h16 + ho; ho += (size_t)NLc * 2 * DINc * Dm;  // 3,145,728
    _Float16* w16xp = h16 + ho; ho += (size_t)NLc * 64 * DINc;      //   196,608
    _Float16* w16dt = h16 + ho; ho += (size_t)NLc * DINc * DTRc;    //    98,304
    _Float16* w16wo = h16 + ho; ho += (size_t)NLc * Dm * DINc;      // 1,572,864
    _Float16* w16ih = h16 + ho; ho += (size_t)2 * 4 * H2c * Dm;     // 1,048,576
    _Float16* w16fw = h16 + ho; ho += (size_t)HIDc * (Dm + 2 * H2c);//   524,288
    _Float16* a16   = h16 + ho; ho += (size_t)BL * DINc;            // staging

    // Aliases used only after the Mamba stack is done:
    float* gproj = xzb;                        // 2*BL*1024 == BL*2048
    float* lfb   = ubuf;                       // BL*256
    float* lbb   = ubuf + (size_t)BL * H2c;    // BL*256
    float* comb  = dtb;                        // BL*1024

    auto cvt = [&](const float* src, int srcStride, _Float16* dst, int K, long total) {
        const long blocks = (total / 4 + 255) / 256;
        cvt_f16_kernel<<<(int)blocks, 256, 0, stream>>>(src, srcStride, dst, K, total);
    };
    auto gemm = [&](const _Float16* A, const _Float16* W,
                    const float* b0, const float* b1,
                    float* C, int M, int N, int K, int act) {
        const int strips = (M / 16) * (N / 64);
        const int blocks = (strips + 7) / 8;
        if (act == ACT_SOFTPLUS)
            gemm_wmma_f16_kernel<ACT_SOFTPLUS><<<blocks, 256, 0, stream>>>(
                A, W, b0, b1, C, M, N, K);
        else
            gemm_wmma_f16_kernel<ACT_NONE><<<blocks, 256, 0, stream>>>(
                A, W, b0, b1, C, M, N, K);
    };

    // ---- one-time (per launch) weight conversions to f16 ----
    cvt(w_in, Dm,   w16in, Dm,   (long)NLc * 2 * DINc * Dm);
    cvt(xpw,  DINc, w16xp, DINc, (long)NLc * 64 * DINc);
    cvt(dtw,  DTRc, w16dt, DTRc, (long)NLc * DINc * DTRc);
    cvt(wo,   DINc, w16wo, DINc, (long)NLc * Dm * DINc);
    cvt(wih,  Dm,   w16ih, Dm,   (long)2 * 4 * H2c * Dm);
    cvt(fw,   Dm + 2 * H2c, w16fw, Dm + 2 * H2c, (long)HIDc * (Dm + 2 * H2c));

    hipMemcpyAsync(mbuf, x, sizeof(float) * (size_t)BL * Dm,
                   hipMemcpyDeviceToDevice, stream);

    for (int i = 0; i < NLc; ++i) {
        // xz = m @ in_proj_w^T                      [2048 x 2048], K=512
        cvt(mbuf, Dm, a16, Dm, (long)BL * Dm);
        gemm(a16, w16in + (size_t)i * 2 * DINc * Dm, nullptr, nullptr,
             xzb, BL, 2 * DINc, Dm, ACT_NONE);
        // u = silu(causal_conv(xin) + cb)
        conv_silu_kernel<<<(BL * DINc + 255) / 256, 256, 0, stream>>>(
            xzb, conv_w + (size_t)i * DINc * DCVc, conv_b + (size_t)i * DINc, ubuf);
        // xdbl = u @ x_proj_w^T                     [2048 x 64], K=1024
        cvt(ubuf, DINc, a16, DINc, (long)BL * DINc);
        gemm(a16, w16xp + (size_t)i * 64 * DINc, nullptr, nullptr,
             xdbl, BL, 64, DINc, ACT_NONE);
        // dt = softplus(dt_in @ dw^T + db)          [2048 x 1024], K=32
        cvt(xdbl, 64, a16, DTRc, (long)BL * DTRc);
        gemm(a16, w16dt + (size_t)i * DINc * DTRc,
             dtbias + (size_t)i * DINc, nullptr, dtb, BL, DINc, DTRc, ACT_SOFTPLUS);
        // selective scan + u*Dp + silu(z) gating
        scan_kernel<<<Bsz * 4, 256, 0, stream>>>(
            dtb, ubuf, xdbl, xzb,
            Alog + (size_t)i * DINc * DSTc, Dp + (size_t)i * DINc, ybuf);
        // mo = y @ out_proj_w^T                     [2048 x 512], K=1024
        cvt(ybuf, DINc, a16, DINc, (long)BL * DINc);
        gemm(a16, w16wo + (size_t)i * Dm * DINc, nullptr, nullptr,
             mob, BL, Dm, DINc, ACT_NONE);
        // m = LayerNorm(m + mo)
        add_ln_kernel<<<BL, 256, 0, stream>>>(mbuf, mob,
                                              lng + (size_t)i * Dm,
                                              lnb + (size_t)i * Dm);
    }

    // LSTM input projections (bias_ih + bias_hh fused): [2048 x 1024], K=512
    cvt(x, Dm, a16, Dm, (long)BL * Dm);
    for (int dir = 0; dir < 2; ++dir) {
        gemm(a16, w16ih + (size_t)dir * 4 * H2c * Dm,
             bih + (size_t)dir * 4 * H2c, bhh + (size_t)dir * 4 * H2c,
             gproj + (size_t)dir * BL * 4 * H2c, BL, 4 * H2c, Dm, ACT_NONE);
    }
    lstm_kernel<<<2, 1024, 0, stream>>>(gproj, whh, lfb, lbb);

    concat_kernel<<<(BL * (Dm + 2 * H2c) + 255) / 256, 256, 0, stream>>>(
        mbuf, lfb, lbb, comb);

    // out = comb @ fusion_w^T + fusion_b           [2048 x 512], K=1024
    cvt(comb, Dm + 2 * H2c, a16, Dm + 2 * H2c, (long)BL * (Dm + 2 * H2c));
    gemm(a16, w16fw, fb, nullptr, (float*)d_out,
         BL, HIDc, Dm + 2 * H2c, ACT_NONE);
}
